// RenderMAEDecoder_7267084665375
// MI455X (gfx1250) — matrix operand order repaired and checked by reading, not compiled, not run on hardware
//
#include <hip/hip_runtime.h>
#include <hip/hip_bf16.h>
#include <float.h>

// ---------------- problem constants ----------------
#define BB    32
#define GG    256
#define NPTS  8192
#define DD    384
#define DEPTH 4
#define HH    6
#define SS    32
#define HDIM  64            // DD / HH
#define TT    (2 * GG)      // 512
#define BT    (BB * TT)     // 16384
#define D4    (4 * DD)      // 1536
#define D3    (3 * DD)      // 1152

typedef __bf16 bf16_t;
typedef __attribute__((ext_vector_type(16))) __bf16 v16bf;
typedef __attribute__((ext_vector_type(8)))  float  v8f;

union Frag { uint4 q[2]; v16bf v; };

__device__ __forceinline__ float gelu_exact(float x) {
    return 0.5f * x * (1.0f + erff(x * 0.70710678118654752f));
}

// ---------------- FPS: one block per batch ----------------
__global__ void __launch_bounds__(1024) fps_kernel(const float* __restrict__ pts,
                                                   float* __restrict__ centers) {
    __shared__ float sdist[NPTS];
    __shared__ float rv[1024];
    __shared__ int   ri[1024];
    __shared__ float scur[3];
    const int b = blockIdx.x;
    const float* p = pts + (size_t)b * NPTS * 3;
    float* cout = centers + (size_t)b * GG * 3;
    const int tid = threadIdx.x;

    for (int i = tid; i < NPTS; i += 1024) sdist[i] = 1e10f;
    if (tid == 0) {
        scur[0] = p[0]; scur[1] = p[1]; scur[2] = p[2];
        cout[0] = p[0]; cout[1] = p[1]; cout[2] = p[2];
    }
    __syncthreads();

    for (int g = 1; g < GG; ++g) {
        const float cx = scur[0], cy = scur[1], cz = scur[2];
        float best = -1.0f; int bidx = NPTS;
        for (int i = tid; i < NPTS; i += 1024) {
            float dx = p[3 * i]     - cx;
            float dy = p[3 * i + 1] - cy;
            float dz = p[3 * i + 2] - cz;
            float d  = dx * dx + dy * dy + dz * dz;
            float nd = fminf(sdist[i], d);
            sdist[i] = nd;
            if (nd > best) { best = nd; bidx = i; }   // strided i ascending -> lowest idx kept
        }
        rv[tid] = best; ri[tid] = bidx;
        __syncthreads();
        for (int s = 512; s > 0; s >>= 1) {
            if (tid < s) {
                float ov = rv[tid + s]; int oi = ri[tid + s];
                if (ov > rv[tid] || (ov == rv[tid] && oi < ri[tid])) { rv[tid] = ov; ri[tid] = oi; }
            }
            __syncthreads();
        }
        if (tid == 0) {
            int ni = ri[0];
            scur[0] = p[3 * ni]; scur[1] = p[3 * ni + 1]; scur[2] = p[3 * ni + 2];
            cout[3 * g] = scur[0]; cout[3 * g + 1] = scur[1]; cout[3 * g + 2] = scur[2];
        }
        __syncthreads();
    }
}

// ---------------- KNN: one block per (b,g); writes mask_group into d_out ----------------
__global__ void __launch_bounds__(256) knn_kernel(const float* __restrict__ pts,
                                                  const float* __restrict__ centers,
                                                  float* __restrict__ mg_out) {
    __shared__ float sdist[NPTS];
    __shared__ float rv[256];
    __shared__ int   ri[256];
    const int bg = blockIdx.x;
    const int b  = bg / GG;
    const float* p = pts + (size_t)b * NPTS * 3;
    const float* c = centers + (size_t)bg * 3;
    const int tid = threadIdx.x;
    const float cx = c[0], cy = c[1], cz = c[2];
    const float cc = cx * cx + cy * cy + cz * cz;
    for (int i = tid; i < NPTS; i += 256) {
        float px = p[3 * i], py = p[3 * i + 1], pz = p[3 * i + 2];
        // same formula as reference: |c|^2 + |p|^2 - 2 c.p
        sdist[i] = cc + (px * px + py * py + pz * pz) - 2.0f * (cx * px + cy * py + cz * pz);
    }
    __syncthreads();
    float* out = mg_out + (size_t)bg * SS * 3;
    for (int s = 0; s < SS; ++s) {
        float best = FLT_MAX; int bidx = NPTS;
        for (int i = tid; i < NPTS; i += 256) {
            float v = sdist[i];
            if (v < best) { best = v; bidx = i; }
        }
        rv[tid] = best; ri[tid] = bidx;
        __syncthreads();
        for (int st = 128; st > 0; st >>= 1) {
            if (tid < st) {
                float ov = rv[tid + st]; int oi = ri[tid + st];
                if (ov < rv[tid] || (ov == rv[tid] && oi < ri[tid])) { rv[tid] = ov; ri[tid] = oi; }
            }
            __syncthreads();
        }
        if (tid == 0) {
            int ni = ri[0];
            sdist[ni] = FLT_MAX;
            out[3 * s]     = p[3 * ni]     - cx;
            out[3 * s + 1] = p[3 * ni + 1] - cy;
            out[3 * s + 2] = p[3 * ni + 2] - cz;
        }
        __syncthreads();
    }
}

// ---------------- positional embedding: one block per (b,t) row ----------------
__global__ void __launch_bounds__(128) pos_embed_kernel(const float* __restrict__ vis_c,
                                                        const float* __restrict__ mask_c,
                                                        const float* __restrict__ w1,
                                                        const float* __restrict__ b1,
                                                        const float* __restrict__ w2,
                                                        const float* __restrict__ b2,
                                                        float* __restrict__ full_pos) {
    __shared__ float hid[128];
    const int row = blockIdx.x;
    const int b = row / TT, t = row % TT;
    const float* c = (t < GG) ? (vis_c + ((size_t)b * GG + t) * 3)
                              : (mask_c + ((size_t)b * GG + (t - GG)) * 3);
    const float c0 = c[0], c1 = c[1], c2 = c[2];
    const int j = threadIdx.x;
    float h = c0 * w1[j] + c1 * w1[128 + j] + c2 * w1[256 + j] + b1[j];
    hid[j] = gelu_exact(h);
    __syncthreads();
    float* outp = full_pos + (size_t)row * DD;
    for (int d = j; d < DD; d += 128) {
        float acc = b2[d];
        for (int k = 0; k < 128; ++k) acc += hid[k] * w2[(size_t)k * DD + d];
        outp[d] = acc;
    }
}

// ---------------- token init ----------------
__global__ void init_x_kernel(const float* __restrict__ vis_token,
                              const float* __restrict__ mask_token,
                              float* __restrict__ x) {
    size_t i = (size_t)blockIdx.x * blockDim.x + threadIdx.x;
    if (i >= (size_t)BT * DD) return;
    size_t row = i / DD; int d = (int)(i % DD);
    int b = (int)(row / TT), t = (int)(row % TT);
    x[i] = (t < GG) ? vis_token[((size_t)b * GG + t) * DD + d] : mask_token[d];
}

// ---------------- x += pos ; h = LN(x) -> bf16 ----------------
__global__ void __launch_bounds__(128) add_pos_ln_kernel(float* __restrict__ x,
                                                         const float* __restrict__ pos,
                                                         const float* __restrict__ g,
                                                         const float* __restrict__ be,
                                                         bf16_t* __restrict__ h) {
    __shared__ float red[128];
    const int row = blockIdx.x, tid = threadIdx.x;
    float* xr = x + (size_t)row * DD;
    const float* pr = pos + (size_t)row * DD;
    float v[3]; float s = 0.f;
#pragma unroll
    for (int i = 0; i < 3; ++i) {
        int d = tid + i * 128;
        float t = xr[d] + pr[d];
        v[i] = t; xr[d] = t; s += t;
    }
    red[tid] = s; __syncthreads();
    for (int st = 64; st > 0; st >>= 1) { if (tid < st) red[tid] += red[tid + st]; __syncthreads(); }
    float mu = red[0] * (1.0f / DD);
    __syncthreads();
    float vs = 0.f;
#pragma unroll
    for (int i = 0; i < 3; ++i) { float d = v[i] - mu; vs += d * d; }
    red[tid] = vs; __syncthreads();
    for (int st = 64; st > 0; st >>= 1) { if (tid < st) red[tid] += red[tid + st]; __syncthreads(); }
    float inv = rsqrtf(red[0] * (1.0f / DD) + 1e-5f);
#pragma unroll
    for (int i = 0; i < 3; ++i) {
        int d = tid + i * 128;
        h[(size_t)row * DD + d] = (bf16_t)((v[i] - mu) * inv * g[d] + be[d]);
    }
}

// ---------------- h = LN(x) -> bf16 (no pos add) ----------------
__global__ void __launch_bounds__(128) ln_kernel(const float* __restrict__ x,
                                                 const float* __restrict__ g,
                                                 const float* __restrict__ be,
                                                 bf16_t* __restrict__ h) {
    __shared__ float red[128];
    const int row = blockIdx.x, tid = threadIdx.x;
    const float* xr = x + (size_t)row * DD;
    float v[3]; float s = 0.f;
#pragma unroll
    for (int i = 0; i < 3; ++i) { v[i] = xr[tid + i * 128]; s += v[i]; }
    red[tid] = s; __syncthreads();
    for (int st = 64; st > 0; st >>= 1) { if (tid < st) red[tid] += red[tid + st]; __syncthreads(); }
    float mu = red[0] * (1.0f / DD);
    __syncthreads();
    float vs = 0.f;
#pragma unroll
    for (int i = 0; i < 3; ++i) { float d = v[i] - mu; vs += d * d; }
    red[tid] = vs; __syncthreads();
    for (int st = 64; st > 0; st >>= 1) { if (tid < st) red[tid] += red[tid + st]; __syncthreads(); }
    float inv = rsqrtf(red[0] * (1.0f / DD) + 1e-5f);
#pragma unroll
    for (int i = 0; i < 3; ++i) {
        int d = tid + i * 128;
        h[(size_t)row * DD + d] = (bf16_t)((v[i] - mu) * inv * g[d] + be[d]);
    }
}

// ---------------- final LN over masked rows only ----------------
__global__ void __launch_bounds__(128) final_ln_kernel(const float* __restrict__ x,
                                                       const float* __restrict__ g,
                                                       const float* __restrict__ be,
                                                       bf16_t* __restrict__ h) {
    __shared__ float red[128];
    const int ro = blockIdx.x, tid = threadIdx.x;       // 0..B*G-1
    const int b = ro / GG, gg = ro % GG;
    const float* xr = x + ((size_t)b * TT + GG + gg) * DD;
    float v[3]; float s = 0.f;
#pragma unroll
    for (int i = 0; i < 3; ++i) { v[i] = xr[tid + i * 128]; s += v[i]; }
    red[tid] = s; __syncthreads();
    for (int st = 64; st > 0; st >>= 1) { if (tid < st) red[tid] += red[tid + st]; __syncthreads(); }
    float mu = red[0] * (1.0f / DD);
    __syncthreads();
    float vs = 0.f;
#pragma unroll
    for (int i = 0; i < 3; ++i) { float d = v[i] - mu; vs += d * d; }
    red[tid] = vs; __syncthreads();
    for (int st = 64; st > 0; st >>= 1) { if (tid < st) red[tid] += red[tid + st]; __syncthreads(); }
    float inv = rsqrtf(red[0] * (1.0f / DD) + 1e-5f);
#pragma unroll
    for (int i = 0; i < 3; ++i) {
        int d = tid + i * 128;
        h[(size_t)ro * DD + d] = (bf16_t)((v[i] - mu) * inv * g[d] + be[d]);
    }
}

// ---------------- weight transpose f32[K,N] -> bf16[N,K] ----------------
__global__ void transpose_bf16_kernel(const float* __restrict__ in, bf16_t* __restrict__ out,
                                      int K, int N) {
    size_t total = (size_t)K * N;
    for (size_t i = (size_t)blockIdx.x * blockDim.x + threadIdx.x; i < total;
         i += (size_t)gridDim.x * blockDim.x) {
        int k = (int)(i / N), n = (int)(i % N);
        out[(size_t)n * K + k] = (bf16_t)in[i];
    }
}

// ---------------- elementwise ----------------
__global__ void residual_add_kernel(float* __restrict__ x, const float* __restrict__ y, size_t n) {
    size_t i = (size_t)blockIdx.x * blockDim.x + threadIdx.x;
    if (i < n) x[i] += y[i];
}

__global__ void gelu_bf16_kernel(const float* __restrict__ in, bf16_t* __restrict__ out, size_t n) {
    size_t i = (size_t)blockIdx.x * blockDim.x + threadIdx.x;
    if (i < n) out[i] = (bf16_t)gelu_exact(in[i]);
}

// ---------------- qkv repack: f32 [BT,1152] -> Q/K [b,h,t,hd] bf16, V^T [b,h,hd,t] bf16 ----
__global__ void repack_qkv_kernel(const float* __restrict__ qkv,
                                  bf16_t* __restrict__ qb, bf16_t* __restrict__ kb,
                                  bf16_t* __restrict__ vtb) {
    size_t i = (size_t)blockIdx.x * blockDim.x + threadIdx.x;
    if (i >= (size_t)BT * DD) return;
    size_t row = i / DD; int hd_all = (int)(i % DD);
    int h = hd_all / HDIM, hd = hd_all % HDIM;
    int b = (int)(row / TT), t = (int)(row % TT);
    const float* r = qkv + row * D3;
    size_t qidx = (((size_t)b * HH + h) * TT + t) * HDIM + hd;
    qb[qidx]  = (bf16_t)r[h * HDIM + hd];
    kb[qidx]  = (bf16_t)r[DD + h * HDIM + hd];
    vtb[(((size_t)b * HH + h) * HDIM + hd) * TT + t] = (bf16_t)r[2 * DD + h * HDIM + hd];
}

// ---------------- attention-out repack: f32 [b,h,t,hd] -> bf16 [bt, D] ----------------
__global__ void repack_o_kernel(const float* __restrict__ o, bf16_t* __restrict__ ob) {
    size_t i = (size_t)blockIdx.x * blockDim.x + threadIdx.x;
    if (i >= (size_t)BT * DD) return;
    size_t row = i / DD; int hd_all = (int)(i % DD);
    int h = hd_all / HDIM, hd = hd_all % HDIM;
    int b = (int)(row / TT), t = (int)(row % TT);
    ob[i] = (bf16_t)o[(((size_t)b * HH + h) * TT + t) * HDIM + hd];
}

// ---------------- softmax over bf16 rows of length TT (in place) ----------------
__global__ void __launch_bounds__(128) softmax_kernel(bf16_t* __restrict__ P) {
    __shared__ float red[128];
    const int row = blockIdx.x, tid = threadIdx.x;
    bf16_t* pr = P + (size_t)row * TT;
    float v[4];
#pragma unroll
    for (int i = 0; i < 4; ++i) v[i] = (float)pr[tid + i * 128];
    float m = fmaxf(fmaxf(v[0], v[1]), fmaxf(v[2], v[3]));
    red[tid] = m; __syncthreads();
    for (int st = 64; st > 0; st >>= 1) { if (tid < st) red[tid] = fmaxf(red[tid], red[tid + st]); __syncthreads(); }
    m = red[0];
    __syncthreads();
    float s = 0.f;
#pragma unroll
    for (int i = 0; i < 4; ++i) { v[i] = __expf(v[i] - m); s += v[i]; }
    red[tid] = s; __syncthreads();
    for (int st = 64; st > 0; st >>= 1) { if (tid < st) red[tid] += red[tid + st]; __syncthreads(); }
    float inv = 1.0f / red[0];
#pragma unroll
    for (int i = 0; i < 4; ++i) pr[tid + i * 128] = (bf16_t)(v[i] * inv);
}

// ---------------- WMMA GEMM: C[M,N] = alpha * A[M,K] x Bt[N,K]^T + bias -----------------
// A, Bt bf16 row-major (Bt is the pre-transposed operand: Bt[n][k]).
// One wave computes a 16(M) x 64(N) tile with v_wmma_f32_16x16x32_bf16.
// Fast path: K-loop unrolled x2 with two fragment register sets that alternate
// by ROLE (no inter-set copies), so loads for one set overlap WMMAs of the
// other and no v_mov shuffling is needed at the loop back-edge.
__global__ void __launch_bounds__(256)
gemm_wmma_kernel(const bf16_t* __restrict__ A, const bf16_t* __restrict__ Bt,
                 float* __restrict__ Cf, bf16_t* __restrict__ Cb,
                 const float* __restrict__ bias, float alpha,
                 int M, int N, int K, int nb, long sA, long sB, long sC) {
    const int lane = threadIdx.x & 31;
    const int wave = threadIdx.x >> 5;
    const int tiles_m = M >> 4;
    const int tiles_n = (N + 63) >> 6;
    const long wpb = (long)tiles_m * tiles_n;
    const long wid = (long)blockIdx.x * 8 + wave;
    if (wid >= wpb * nb) return;                  // whole-wave uniform: EXEC all-1 at WMMA
    const int batch = (int)(wid / wpb);
    const int rem   = (int)(wid % wpb);
    const int m0 = (rem % tiles_m) << 4;
    const int n0 = (rem / tiles_m) << 6;
    const int mr   = lane & 15;
    const int half = lane >> 4;
    const bf16_t* ap = A  + (long)batch * sA + (long)(m0 + mr) * K + 16 * half;
    const bf16_t* bp = Bt + (long)batch * sB + (long)(n0 + mr) * K + 16 * half;

    v8f c0 = {}, c1 = {}, c2 = {}, c3 = {};
    int nsub;

    if (n0 + 64 <= N && (K & 63) == 0) {
        nsub = 4;
        // fragment set 0, loaded at k = 0
        Frag a0, p00, p01, p02, p03;
        a0.q[0]  = *reinterpret_cast<const uint4*>(ap);
        a0.q[1]  = *reinterpret_cast<const uint4*>(ap + 8);
        p00.q[0] = *reinterpret_cast<const uint4*>(bp);
        p00.q[1] = *reinterpret_cast<const uint4*>(bp + 8);
        p01.q[0] = *reinterpret_cast<const uint4*>(bp + 16L * K);
        p01.q[1] = *reinterpret_cast<const uint4*>(bp + 16L * K + 8);
        p02.q[0] = *reinterpret_cast<const uint4*>(bp + 32L * K);
        p02.q[1] = *reinterpret_cast<const uint4*>(bp + 32L * K + 8);
        p03.q[0] = *reinterpret_cast<const uint4*>(bp + 48L * K);
        p03.q[1] = *reinterpret_cast<const uint4*>(bp + 48L * K + 8);
        int k0 = 0;
        while (true) {
            // load set 1 at k0+32 (always in range: K % 64 == 0, k0 < K)
            Frag a1, p10, p11, p12, p13;
            {
                const bf16_t* ap2 = ap + k0 + 32;
                const bf16_t* bp2 = bp + k0 + 32;
                a1.q[0]  = *reinterpret_cast<const uint4*>(ap2);
                a1.q[1]  = *reinterpret_cast<const uint4*>(ap2 + 8);
                p10.q[0] = *reinterpret_cast<const uint4*>(bp2);
                p10.q[1] = *reinterpret_cast<const uint4*>(bp2 + 8);
                p11.q[0] = *reinterpret_cast<const uint4*>(bp2 + 16L * K);
                p11.q[1] = *reinterpret_cast<const uint4*>(bp2 + 16L * K + 8);
                p12.q[0] = *reinterpret_cast<const uint4*>(bp2 + 32L * K);
                p12.q[1] = *reinterpret_cast<const uint4*>(bp2 + 32L * K + 8);
                p13.q[0] = *reinterpret_cast<const uint4*>(bp2 + 48L * K);
                p13.q[1] = *reinterpret_cast<const uint4*>(bp2 + 48L * K + 8);
            }
            // compute with set 0 (loads for set 1 in flight)
            c0 = __builtin_amdgcn_wmma_f32_16x16x32_bf16(false, a0.v, false, p00.v, (short)0, c0, false, false);
            c1 = __builtin_amdgcn_wmma_f32_16x16x32_bf16(false, a0.v, false, p01.v, (short)0, c1, false, false);
            c2 = __builtin_amdgcn_wmma_f32_16x16x32_bf16(false, a0.v, false, p02.v, (short)0, c2, false, false);
            c3 = __builtin_amdgcn_wmma_f32_16x16x32_bf16(false, a0.v, false, p03.v, (short)0, c3, false, false);
            k0 += 64;
            if (k0 < K) {
                // reload set 0 at k0 (next pair), compute with set 1
                const bf16_t* ap3 = ap + k0;
                const bf16_t* bp3 = bp + k0;
                a0.q[0]  = *reinterpret_cast<const uint4*>(ap3);
                a0.q[1]  = *reinterpret_cast<const uint4*>(ap3 + 8);
                p00.q[0] = *reinterpret_cast<const uint4*>(bp3);
                p00.q[1] = *reinterpret_cast<const uint4*>(bp3 + 8);
                p01.q[0] = *reinterpret_cast<const uint4*>(bp3 + 16L * K);
                p01.q[1] = *reinterpret_cast<const uint4*>(bp3 + 16L * K + 8);
                p02.q[0] = *reinterpret_cast<const uint4*>(bp3 + 32L * K);
                p02.q[1] = *reinterpret_cast<const uint4*>(bp3 + 32L * K + 8);
                p03.q[0] = *reinterpret_cast<const uint4*>(bp3 + 48L * K);
                p03.q[1] = *reinterpret_cast<const uint4*>(bp3 + 48L * K + 8);
                c0 = __builtin_amdgcn_wmma_f32_16x16x32_bf16(false, a1.v, false, p10.v, (short)0, c0, false, false);
                c1 = __builtin_amdgcn_wmma_f32_16x16x32_bf16(false, a1.v, false, p11.v, (short)0, c1, false, false);
                c2 = __builtin_amdgcn_wmma_f32_16x16x32_bf16(false, a1.v, false, p12.v, (short)0, c2, false, false);
                c3 = __builtin_amdgcn_wmma_f32_16x16x32_bf16(false, a1.v, false, p13.v, (short)0, c3, false, false);
            } else {
                c0 = __builtin_amdgcn_wmma_f32_16x16x32_bf16(false, a1.v, false, p10.v, (short)0, c0, false, false);
                c1 = __builtin_amdgcn_wmma_f32_16x16x32_bf16(false, a1.v, false, p11.v, (short)0, c1, false, false);
                c2 = __builtin_amdgcn_wmma_f32_16x16x32_bf16(false, a1.v, false, p12.v, (short)0, c2, false, false);
                c3 = __builtin_amdgcn_wmma_f32_16x16x32_bf16(false, a1.v, false, p13.v, (short)0, c3, false, false);
                break;
            }
        }
    } else {
        // partial N tile or odd K: simple guarded loop (head GEMM remainder)
        const int nsub0 = (N - n0 + 15) >> 4;
        nsub = nsub0 < 4 ? nsub0 : 4;
        for (int k0 = 0; k0 < K; k0 += 32) {
            Frag a;
            a.q[0] = *reinterpret_cast<const uint4*>(ap + k0);
            a.q[1] = *reinterpret_cast<const uint4*>(ap + k0 + 8);
            if (k0 + 32 < K) __builtin_prefetch(ap + k0 + 32, 0, 3);
#pragma unroll
            for (int i = 0; i < 4; ++i) {
                if (i < nsub) {
                    Frag bf;
                    const bf16_t* bpp = bp + (long)i * 16 * K + k0;
                    bf.q[0] = *reinterpret_cast<const uint4*>(bpp);
                    bf.q[1] = *reinterpret_cast<const uint4*>(bpp + 8);
                    v8f acc = (i == 0) ? c0 : (i == 1) ? c1 : (i == 2) ? c2 : c3;
                    acc = __builtin_amdgcn_wmma_f32_16x16x32_bf16(
                              false, a.v, false, bf.v, (short)0, acc, false, false);
                    if (i == 0) c0 = acc; else if (i == 1) c1 = acc; else if (i == 2) c2 = acc; else c3 = acc;
                }
            }
        }
    }
#pragma unroll
    for (int i = 0; i < 4; ++i) {
        if (i < nsub) {
            v8f acc = (i == 0) ? c0 : (i == 1) ? c1 : (i == 2) ? c2 : c3;
            const int col = n0 + i * 16 + mr;
            const float bv = bias ? bias[col] : 0.0f;
#pragma unroll
            for (int r = 0; r < 8; ++r) {
                const int rowm = m0 + r + 8 * half;
                const float val = alpha * acc[r] + bv;
                const long off = (long)batch * sC + (long)rowm * N + col;
                if (Cf) Cf[off] = val; else Cb[off] = (bf16_t)val;
            }
        }
    }
}

// ---------------- host side ----------------
static inline void launch_gemm(const bf16_t* A, const bf16_t* Bt, float* Cf, bf16_t* Cb,
                               const float* bias, float alpha, int M, int N, int K, int nb,
                               long sA, long sB, long sC, hipStream_t st) {
    long waves = (long)(M / 16) * ((N + 63) / 64) * nb;
    int blocks = (int)((waves + 7) / 8);
    gemm_wmma_kernel<<<blocks, 256, 0, st>>>(A, Bt, Cf, Cb, bias, alpha, M, N, K, nb, sA, sB, sC);
}

extern "C" void kernel_launch(void* const* d_in, const int* in_sizes, int n_in,
                              void* d_out, int out_size, void* d_ws, size_t ws_size,
                              hipStream_t stream) {
    (void)in_sizes; (void)n_in; (void)out_size; (void)ws_size;
    const float* vis_token = (const float*)d_in[0];
    const float* vis_cent  = (const float*)d_in[1];
    const float* mask_pts  = (const float*)d_in[2];
    const float* mask_tok  = (const float*)d_in[3];
    const float* pos_w1    = (const float*)d_in[4];
    const float* pos_b1    = (const float*)d_in[5];
    const float* pos_w2    = (const float*)d_in[6];
    const float* pos_b2    = (const float*)d_in[7];
    const float* ln1_g = (const float*)d_in[8],  *ln1_b = (const float*)d_in[9];
    const float* qkv_w = (const float*)d_in[10], *qkv_b = (const float*)d_in[11];
    const float* prj_w = (const float*)d_in[12], *prj_b = (const float*)d_in[13];
    const float* ln2_g = (const float*)d_in[14], *ln2_b = (const float*)d_in[15];
    const float* fc1_w = (const float*)d_in[16], *fc1_b = (const float*)d_in[17];
    const float* fc2_w = (const float*)d_in[18], *fc2_b = (const float*)d_in[19];
    const float* nrm_g = (const float*)d_in[20], *nrm_b = (const float*)d_in[21];
    const float* hd_w  = (const float*)d_in[22], *hd_b  = (const float*)d_in[23];
    float* out = (float*)d_out;                       // [pred | mask_group]
    float* mg_out = out + (size_t)BB * GG * SS * 3;

    char* ws = (char*)d_ws;
    float*  cent  = (float*)(ws);                         // B*G*3
    float*  x     = (float*)(ws + 98304);                 // BT*D f32
    float*  fpos  = (float*)(ws + 25264128);              // BT*D f32
    bf16_t* hbuf  = (bf16_t*)(ws + 50429952);             // up to BT*4D bf16
    float*  gbuf  = (float*)(ws + 100761600);             // up to BT*4D f32  (shares with P)
    bf16_t* Pbuf  = (bf16_t*)gbuf;                        // B*H*T*T bf16 (same 100.7MB)
    float*  obuf  = (float*)(ws + 201424896);             // BT*D f32
    bf16_t* qb    = (bf16_t*)(ws + 226590720);            // B*H*T*HD
    bf16_t* kb    = (bf16_t*)(ws + 239173632);
    bf16_t* vtb   = (bf16_t*)(ws + 251756544);
    bf16_t* wqT   = (bf16_t*)(ws + 264339456);            // 4 x [1152,384]
    bf16_t* wpT   = (bf16_t*)(ws + 267878400);            // 4 x [384,384]
    bf16_t* w1T   = (bf16_t*)(ws + 269058048);            // 4 x [1536,384]
    bf16_t* w2T   = (bf16_t*)(ws + 273776640);            // 4 x [384,1536]
    bf16_t* whT   = (bf16_t*)(ws + 278495232);            // [96,384]

    // ---- weights -> bf16, transposed to [N,K]
    for (int d = 0; d < DEPTH; ++d) {
        transpose_bf16_kernel<<<1728, 256, 0, stream>>>(qkv_w + (size_t)d * DD * D3, wqT + (size_t)d * D3 * DD, DD, D3);
        transpose_bf16_kernel<<<576,  256, 0, stream>>>(prj_w + (size_t)d * DD * DD, wpT + (size_t)d * DD * DD, DD, DD);
        transpose_bf16_kernel<<<2304, 256, 0, stream>>>(fc1_w + (size_t)d * DD * D4, w1T + (size_t)d * D4 * DD, DD, D4);
        transpose_bf16_kernel<<<2304, 256, 0, stream>>>(fc2_w + (size_t)d * D4 * DD, w2T + (size_t)d * DD * D4, D4, DD);
    }
    transpose_bf16_kernel<<<144, 256, 0, stream>>>(hd_w, whT, DD, 3 * SS);

    // ---- geometry
    fps_kernel<<<BB, 1024, 0, stream>>>(mask_pts, cent);
    knn_kernel<<<BB * GG, 256, 0, stream>>>(mask_pts, cent, mg_out);
    pos_embed_kernel<<<BT, 128, 0, stream>>>(vis_cent, cent, pos_w1, pos_b1, pos_w2, pos_b2, fpos);
    init_x_kernel<<<(BT * DD + 255) / 256, 256, 0, stream>>>(vis_token, mask_tok, x);

    const size_t nBTD = (size_t)BT * DD;
    const float  scale = 0.125f;   // 1/sqrt(64)

    for (int d = 0; d < DEPTH; ++d) {
        // x += pos ; h = LN1(x)
        add_pos_ln_kernel<<<BT, 128, 0, stream>>>(x, fpos, ln1_g + d * DD, ln1_b + d * DD, hbuf);
        // qkv = h @ Wqkv + b
        launch_gemm(hbuf, wqT + (size_t)d * D3 * DD, gbuf, nullptr, qkv_b + d * D3, 1.0f,
                    BT, D3, DD, 1, 0, 0, 0, stream);
        repack_qkv_kernel<<<(int)((nBTD + 255) / 256), 256, 0, stream>>>(gbuf, qb, kb, vtb);
        // scores = scale * Q K^T  (bf16 out, per (b,h))
        launch_gemm(qb, kb, nullptr, Pbuf, nullptr, scale, TT, TT, HDIM, BB * HH,
                    (long)TT * HDIM, (long)TT * HDIM, (long)TT * TT, stream);
        softmax_kernel<<<BB * HH * TT, 128, 0, stream>>>(Pbuf);
        // O = P V (f32 out, per (b,h))
        launch_gemm(Pbuf, vtb, obuf, nullptr, nullptr, 1.0f, TT, HDIM, TT, BB * HH,
                    (long)TT * TT, (long)HDIM * TT, (long)TT * HDIM, stream);
        repack_o_kernel<<<(int)((nBTD + 255) / 256), 256, 0, stream>>>(obuf, hbuf);
        // x += O @ Wproj + b
        launch_gemm(hbuf, wpT + (size_t)d * DD * DD, obuf, nullptr, prj_b + d * DD, 1.0f,
                    BT, DD, DD, 1, 0, 0, 0, stream);
        residual_add_kernel<<<(int)((nBTD + 255) / 256), 256, 0, stream>>>(x, obuf, nBTD);
        // h = LN2(x) ; x += gelu(h @ W1 + b1) @ W2 + b2
        ln_kernel<<<BT, 128, 0, stream>>>(x, ln2_g + d * DD, ln2_b + d * DD, hbuf);
        launch_gemm(hbuf, w1T + (size_t)d * D4 * DD, gbuf, nullptr, fc1_b + d * D4, 1.0f,
                    BT, D4, DD, 1, 0, 0, 0, stream);
        gelu_bf16_kernel<<<(int)(((size_t)BT * D4 + 255) / 256), 256, 0, stream>>>(gbuf, hbuf, (size_t)BT * D4);
        launch_gemm(hbuf, w2T + (size_t)d * DD * D4, obuf, nullptr, fc2_b + d * DD, 1.0f,
                    BT, DD, D4, 1, 0, 0, 0, stream);
        residual_add_kernel<<<(int)((nBTD + 255) / 256), 256, 0, stream>>>(x, obuf, nBTD);
    }

    // final LN (masked rows) + head -> pred_group directly into d_out
    final_ln_kernel<<<BB * GG, 128, 0, stream>>>(x, nrm_g, nrm_b, hbuf);
    launch_gemm(hbuf, whT, out, nullptr, hd_b, 1.0f, BB * GG, 3 * SS, DD, 1, 0, 0, 0, stream);
}
